// DecoderUnit_50843822850103
// MI455X (gfx1250) — matrix operand order repaired and checked by reading, not compiled, not run on hardware
//
#include <hip/hip_runtime.h>

// ---------------------------------------------------------------------------
// Transformer decoder layer for MI455X (gfx1250): bf16 WMMA everywhere,
// async global->LDS copies (ASYNCcnt), DPP16 softmax reductions.
// B=2, S=2048, D=512, H=8, HD=64, EXP=4  ->  BS=4096, FF=2048.
// ---------------------------------------------------------------------------

typedef unsigned short u16;
typedef __bf16  bf16_t;
typedef __attribute__((ext_vector_type(16))) bf16_t v16bf;
typedef __attribute__((ext_vector_type(8)))  float  v8f;

#define BSZ   2
#define SEQ   2048
#define DIM   512
#define NH    8
#define HDM   64
#define FFD   2048
#define ROWS  (BSZ * SEQ)          /* 4096 */

union AFrag { v16bf v; unsigned u[8]; };

static __device__ __forceinline__ u16 f2bf(float x) {
    union { float f; unsigned u; } c; c.f = x;
    unsigned r = (c.u + 0x7fffu + ((c.u >> 16) & 1u)) >> 16;
    return (u16)r;
}

// Async copy of 16 bytes global -> LDS, tracked with ASYNCcnt.
static __device__ __forceinline__ void async_copy16(unsigned ldsAddr,
                                                    const void* gbase,
                                                    int byteOff) {
    asm volatile("global_load_async_to_lds_b128 %0, %1, %2"
                 :: "v"(ldsAddr), "v"(byteOff),
                    "s"((unsigned long long)(uintptr_t)gbase)
                 : "memory");
}
static __device__ __forceinline__ void async_wait0() {
    asm volatile("s_wait_asynccnt 0x0" ::: "memory");
}

// DPP16 butterfly reductions across a 16-lane group (pure VALU, no LDS).
template<int CTRL>
static __device__ __forceinline__ float dppf(float x) {
    union { float f; int i; } a, r;
    a.f = x;
    r.i = __builtin_amdgcn_update_dpp(a.i, a.i, CTRL, 0xf, 0xf, true);
    return r.f;
}
static __device__ __forceinline__ float redmax16(float x) {
    x = fmaxf(x, dppf<0xB1>(x));    // quad_perm xor1
    x = fmaxf(x, dppf<0x4E>(x));    // quad_perm xor2
    x = fmaxf(x, dppf<0x141>(x));   // row_half_mirror (~xor4)
    x = fmaxf(x, dppf<0x140>(x));   // row_mirror (~xor8)
    return x;
}
static __device__ __forceinline__ float redsum16(float x) {
    x += dppf<0xB1>(x);
    x += dppf<0x4E>(x);
    x += dppf<0x141>(x);
    x += dppf<0x140>(x);
    return x;
}

// ------------------------------ converters --------------------------------

__global__ void cvt_f32_bf16(const float* __restrict__ in, u16* __restrict__ out, int n) {
    int i = blockIdx.x * 256 + threadIdx.x;
    if (i < n) out[i] = f2bf(in[i]);
}

// W [H, D, HD] fp32  ->  W' [D, H*HD] bf16   (W'[d, h*HD+e] = W[h,d,e])
__global__ void repack_qkv_w(const float* __restrict__ W, u16* __restrict__ out) {
    int i = blockIdx.x * 256 + threadIdx.x;           // i over H*D*HD = 262144
    if (i >= NH * DIM * HDM) return;
    int h = i / (DIM * HDM);
    int rem = i % (DIM * HDM);
    int d = rem / HDM, e = rem % HDM;
    out[d * DIM + h * HDM + e] = f2bf(W[i]);
}

// ------------------------------ GEMM (bf16 WMMA) ---------------------------
// C[M,N] = A[M,K](bf16) * B[K,N](bf16) + bias, compile-time epilogue.
// Block: 128 thr (4 waves), tile 128x64, K-step 32, 8 WMMA / wave / K-step.
template<bool RELU, bool RES, bool OUTF, bool OUTB>
__global__ __launch_bounds__(128)
void gemm_bf16(const u16* __restrict__ A, const u16* __restrict__ Bm,
               const float* __restrict__ bias, const float* __restrict__ resid,
               float* __restrict__ outF, u16* __restrict__ outB,
               int M, int N, int K) {
    __shared__ __align__(16) u16 As[128 * 32];  // [m][k]
    __shared__ __align__(16) u16 Bs[64 * 32];   // transposed: [n][k]

    const int t = threadIdx.x;
    const int w = t >> 5, lane = t & 31;
    const int half = lane >> 4, l16 = lane & 15;
    const int m0 = blockIdx.y * 128, n0 = blockIdx.x * 64;

    __builtin_assume(K >= 32);

    // A-tile async-copy slots: 512 x 16B chunks, 4 per thread.
    unsigned aLds[4]; int aOff[4];               // element offsets (fixed part)
    #pragma unroll
    for (int i = 0; i < 4; ++i) {
        int c = t + i * 128;
        int r = c >> 2, cc = (c & 3) * 8;
        aLds[i] = (unsigned)(uintptr_t)(&As[r * 32 + cc]);
        aOff[i] = (m0 + r) * K + cc;
    }
    // B-tile (transposed) loader slots: 256 chunks, 2 per thread.
    int bRow[2], bCol[2];
    #pragma unroll
    for (int i = 0; i < 2; ++i) {
        int c = t + i * 128;
        bRow[i] = c >> 3; bCol[i] = (c & 7) * 8;
    }

    v8f acc[2][4] = {};

    for (int kk = 0; kk < K; kk += 32) {
        __syncthreads();
        // A tile: async global->LDS (no VGPR round trip)
        #pragma unroll
        for (int i = 0; i < 4; ++i)
            async_copy16(aLds[i], A, (aOff[i] + kk) * 2);
        // B tile 32x64 -> transposed into Bs[n][k] (VGPR path, overlaps async)
        #pragma unroll
        for (int i = 0; i < 2; ++i) {
            u16 tmp[8];
            *(float4*)tmp = *(const float4*)(Bm + (size_t)(kk + bRow[i]) * N + n0 + bCol[i]);
            #pragma unroll
            for (int j = 0; j < 8; ++j) Bs[(bCol[i] + j) * 32 + bRow[i]] = tmp[j];
        }
        // prefetch next K-step tiles into cache while this one computes
        if (kk + 32 < K) {
            __builtin_prefetch(A + aOff[0] + kk + 32, 0, 1);
            __builtin_prefetch(Bm + (size_t)(kk + 32 + bRow[0]) * N + n0 + bCol[0], 0, 1);
        }
        async_wait0();
        __syncthreads();

        // A fragments: two 16x32 tiles (wave rows w*32 .. w*32+31)
        AFrag a[2];
        #pragma unroll
        for (int sub = 0; sub < 2; ++sub) {
            const u16* ap = &As[(w * 32 + sub * 16 + l16) * 32];
            #pragma unroll
            for (int j2 = 0; j2 < 8; ++j2) {
                int jj = j2 * 2;
                int kidx = half * 16 + jj;
                a[sub].u[j2] = *(const unsigned*)(ap + kidx);
            }
        }
        #pragma unroll
        for (int nt = 0; nt < 4; ++nt) {
            AFrag b0;
            const u16* bp = &Bs[(nt * 16 + l16) * 32 + half * 16];
            #pragma unroll
            for (int j2 = 0; j2 < 8; ++j2) b0.u[j2] = *(const unsigned*)(bp + j2 * 2);
            acc[0][nt] = __builtin_amdgcn_wmma_f32_16x16x32_bf16(
                false, a[0].v, false, b0.v, (short)0, acc[0][nt], false, false);
            acc[1][nt] = __builtin_amdgcn_wmma_f32_16x16x32_bf16(
                false, a[1].v, false, b0.v, (short)0, acc[1][nt], false, false);
        }
    }

    #pragma unroll
    for (int sub = 0; sub < 2; ++sub) {
        const int rowbase = m0 + w * 32 + sub * 16 + half * 8;
        #pragma unroll
        for (int nt = 0; nt < 4; ++nt) {
            const int col = n0 + nt * 16 + l16;
            const float bv = bias[col];
            #pragma unroll
            for (int r = 0; r < 8; ++r) {
                const size_t idx = (size_t)(rowbase + r) * N + col;
                float v = acc[sub][nt][r] + bv;
                if (RELU) v = v > 0.f ? v : 0.f;
                if (RES)  v += resid[idx];
                if (OUTF) outF[idx] = v;
                if (OUTB) outB[idx] = f2bf(v);
            }
        }
    }
}

// --------------------------- flash attention -------------------------------
// q,k,v,out: bf16 [B,S,H*HD] (row-major, row stride D). One block = 64 query
// rows of one (b,h); 4 waves x 16 rows. Key blocks of 32, online softmax.
__global__ __launch_bounds__(128)
void flash_attn(const u16* __restrict__ q, const u16* __restrict__ k,
                const u16* __restrict__ v, const int* __restrict__ mask,
                u16* __restrict__ out, float scale, int hasMask) {
    __shared__ __align__(16) u16 Ks[32 * 64];     // [key][e]
    __shared__ __align__(16) u16 Vs[64 * 32];     // transposed [e][key]
    __shared__ __align__(16) u16 Ps[4][16 * 32];  // per-wave P tile

    const int t = threadIdx.x;
    const int w = t >> 5, lane = t & 31;
    const int half = lane >> 4, l16 = lane & 15;
    const int bh = blockIdx.y;
    const int b = bh >> 3, h = bh & 7;
    const int q0 = blockIdx.x * 64 + w * 16;
    const size_t base = (size_t)b * SEQ * DIM + h * HDM;

    // Q fragments for e=[0,32) and e=[32,64)
    AFrag qa[2];
    {
        const u16* qp = q + base + (size_t)(q0 + l16) * DIM;
        #pragma unroll
        for (int e0 = 0; e0 < 2; ++e0)
            #pragma unroll
            for (int j2 = 0; j2 < 8; ++j2) {
                int jj = j2 * 2;
                int kidx = half * 16 + jj;
                qa[e0].u[j2] = *(const unsigned*)(qp + e0 * 32 + kidx);
            }
    }

    // K-tile async-copy slots (256 chunks, 2 per thread) + V loader slots.
    unsigned kLds[2]; int kFix[2], vRow[2], vCol[2];
    #pragma unroll
    for (int i = 0; i < 2; ++i) {
        int c = t + i * 128;
        int r = c >> 3, cc = (c & 7) * 8;
        kLds[i] = (unsigned)(uintptr_t)(&Ks[r * 64 + cc]);
        kFix[i] = (int)((base + (size_t)r * DIM + cc) * 2);
        vRow[i] = r; vCol[i] = cc;
    }

    v8f o[4] = {};
    float m[8], l[8];
    #pragma unroll
    for (int r = 0; r < 8; ++r) { m[r] = -1e30f; l[r] = 0.f; }

    for (int kt = 0; kt < SEQ / 32; ++kt) {
        const int c0 = kt * 32;
        __syncthreads();
        // K tile: async global->LDS; V tile: load + transpose (overlapped)
        #pragma unroll
        for (int i = 0; i < 2; ++i)
            async_copy16(kLds[i], k, kFix[i] + c0 * DIM * 2);
        #pragma unroll
        for (int i = 0; i < 2; ++i) {
            u16 tmp[8];
            *(float4*)tmp = *(const float4*)(v + base + (size_t)(c0 + vRow[i]) * DIM + vCol[i]);
            #pragma unroll
            for (int j = 0; j < 8; ++j) Vs[(vCol[i] + j) * 32 + vRow[i]] = tmp[j];
        }
        // prefetch next tiles
        if (kt + 1 < SEQ / 32) {
            __builtin_prefetch(k + base + (size_t)(c0 + 32 + vRow[0]) * DIM + vCol[0], 0, 1);
            __builtin_prefetch(v + base + (size_t)(c0 + 32 + vRow[0]) * DIM + vCol[0], 0, 1);
        }
        async_wait0();
        __syncthreads();

        // scores S[16,32] = Q * K^T  (2 col tiles, reduction over 64 in 2 steps)
        v8f s0 = {}, s1 = {};
        #pragma unroll
        for (int e0 = 0; e0 < 2; ++e0) {
            AFrag b0, b1;
            const u16* kp0 = &Ks[l16 * 64 + e0 * 32 + half * 16];
            const u16* kp1 = &Ks[(16 + l16) * 64 + e0 * 32 + half * 16];
            #pragma unroll
            for (int j2 = 0; j2 < 8; ++j2) {
                b0.u[j2] = *(const unsigned*)(kp0 + j2 * 2);
                b1.u[j2] = *(const unsigned*)(kp1 + j2 * 2);
            }
            s0 = __builtin_amdgcn_wmma_f32_16x16x32_bf16(false, qa[e0].v, false, b0.v, (short)0, s0, false, false);
            s1 = __builtin_amdgcn_wmma_f32_16x16x32_bf16(false, qa[e0].v, false, b1.v, (short)0, s1, false, false);
        }

        float sc0[8], sc1[8];
        #pragma unroll
        for (int r = 0; r < 8; ++r) { sc0[r] = s0[r] * scale; sc1[r] = s1[r] * scale; }
        if (hasMask) {
            const int qr = q0 + half * 8;
            #pragma unroll
            for (int r = 0; r < 8; ++r) {
                const int* mrow = mask + (size_t)(qr + r) * SEQ + c0;
                if (mrow[l16] == 0)      sc0[r] = -1e20f;
                if (mrow[16 + l16] == 0) sc1[r] = -1e20f;
            }
        }
        // online softmax: DPP16 butterfly reductions (row lives in a 16-lane group)
        #pragma unroll
        for (int r = 0; r < 8; ++r) {
            float rm = redmax16(fmaxf(sc0[r], sc1[r]));
            float mn = fmaxf(m[r], rm);
            float alpha = __expf(m[r] - mn);
            float p0 = __expf(sc0[r] - mn);
            float p1 = __expf(sc1[r] - mn);
            float rs = redsum16(p0 + p1);
            l[r] = l[r] * alpha + rs;
            m[r] = mn;
            o[0][r] *= alpha; o[1][r] *= alpha; o[2][r] *= alpha; o[3][r] *= alpha;
            Ps[w][(half * 8 + r) * 32 + l16]      = f2bf(p0);
            Ps[w][(half * 8 + r) * 32 + 16 + l16] = f2bf(p1);
        }
        __syncthreads();   // C-layout -> A-layout reshape through LDS

        // O += P[16,32] * V[32,64]
        AFrag pa;
        {
            const u16* pp = &Ps[w][l16 * 32];
            #pragma unroll
            for (int j2 = 0; j2 < 8; ++j2) {
                int jj = j2 * 2;
                int kidx = half * 16 + jj;
                pa.u[j2] = *(const unsigned*)(pp + kidx);
            }
        }
        #pragma unroll
        for (int nt = 0; nt < 4; ++nt) {
            AFrag bv;
            const u16* vp = &Vs[(nt * 16 + l16) * 32 + half * 16];
            #pragma unroll
            for (int j2 = 0; j2 < 8; ++j2) bv.u[j2] = *(const unsigned*)(vp + j2 * 2);
            o[nt] = __builtin_amdgcn_wmma_f32_16x16x32_bf16(
                false, pa.v, false, bv.v, (short)0, o[nt], false, false);
        }
    }

    #pragma unroll
    for (int nt = 0; nt < 4; ++nt)
        #pragma unroll
        for (int r = 0; r < 8; ++r) {
            const int row = q0 + half * 8 + r;
            const int e = nt * 16 + l16;
            out[base + (size_t)row * DIM + e] = f2bf(o[nt][r] / l[r]);
        }
}

// ------------------------------ layernorm ----------------------------------
__global__ __launch_bounds__(128)
void layernorm_k(const float* __restrict__ in, const float* __restrict__ g,
                 const float* __restrict__ bta, float* __restrict__ outF,
                 u16* __restrict__ outB) {
    const int row = blockIdx.x;
    const float* x = in + (size_t)row * DIM;
    const int t = threadIdx.x, w = t >> 5, lane = t & 31;
    float s = 0.f, s2 = 0.f, vals[4];
    #pragma unroll
    for (int i = 0; i < 4; ++i) {
        float v = x[t + i * 128];
        vals[i] = v; s += v; s2 += v * v;
    }
    #pragma unroll
    for (int off = 16; off > 0; off >>= 1) {
        s  += __shfl_xor(s, off, 32);
        s2 += __shfl_xor(s2, off, 32);
    }
    __shared__ float rs[4], rs2[4];
    if (lane == 0) { rs[w] = s; rs2[w] = s2; }
    __syncthreads();
    s  = rs[0] + rs[1] + rs[2] + rs[3];
    s2 = rs2[0] + rs2[1] + rs2[2] + rs2[3];
    const float mean = s * (1.f / DIM);
    const float var  = s2 * (1.f / DIM) - mean * mean;
    const float inv  = rsqrtf(var + 1e-5f);
    #pragma unroll
    for (int i = 0; i < 4; ++i) {
        const int c = t + i * 128;
        const float v = (vals[i] - mean) * inv * g[c] + bta[c];
        outF[(size_t)row * DIM + c] = v;
        if (outB) outB[(size_t)row * DIM + c] = f2bf(v);
    }
}

// ------------------------------ launcher -----------------------------------

extern "C" void kernel_launch(void* const* d_in, const int* in_sizes, int n_in,
                              void* d_out, int out_size, void* d_ws, size_t ws_size,
                              hipStream_t stream) {
    (void)in_sizes; (void)n_in; (void)out_size; (void)ws_size;

    const float* x     = (const float*)d_in[0];
    // d_in[1] = enc_output: unused (matches reference bug)
    const int*   tmask = (const int*)d_in[2];
    const float* Wq1 = (const float*)d_in[3],  *bq1 = (const float*)d_in[4];
    const float* Wk1 = (const float*)d_in[5],  *bk1 = (const float*)d_in[6];
    const float* Wv1 = (const float*)d_in[7],  *bv1 = (const float*)d_in[8];
    const float* Wo1 = (const float*)d_in[9],  *bo1 = (const float*)d_in[10];
    const float* Wq2 = (const float*)d_in[11], *bq2 = (const float*)d_in[12];
    const float* Wk2 = (const float*)d_in[13], *bk2 = (const float*)d_in[14];
    const float* Wv2 = (const float*)d_in[15], *bv2 = (const float*)d_in[16];
    const float* Wo2 = (const float*)d_in[17], *bo2 = (const float*)d_in[18];
    const float* g1 = (const float*)d_in[19], *b1 = (const float*)d_in[20];
    const float* g2 = (const float*)d_in[21], *b2 = (const float*)d_in[22];
    const float* g3 = (const float*)d_in[23], *b3 = (const float*)d_in[24];
    const float* Wf1 = (const float*)d_in[25], *bf1 = (const float*)d_in[26];
    const float* Wf2 = (const float*)d_in[27], *bf2 = (const float*)d_in[28];

    char* ws = (char*)d_ws;
    size_t off = 0;
    auto give = [&](size_t bytes) -> char* {
        char* p = ws + off;
        off += (bytes + 255) & ~(size_t)255;
        return p;
    };

    const size_t ACT_B  = (size_t)ROWS * DIM * 2;   // bf16 activation [4096,512]
    const size_t ACT_F  = (size_t)ROWS * DIM * 4;   // fp32 activation
    const size_t WSQ_B  = (size_t)DIM * DIM * 2;    // bf16 [512,512] weight
    const size_t WFF_B  = (size_t)DIM * FFD * 2;    // bf16 [512,2048] weight
    const size_t HID_B  = (size_t)ROWS * FFD * 2;   // bf16 hidden [4096,2048]

    u16* xb   = (u16*)give(ACT_B);
    u16* wq1p = (u16*)give(WSQ_B); u16* wk1p = (u16*)give(WSQ_B); u16* wv1p = (u16*)give(WSQ_B);
    u16* wq2p = (u16*)give(WSQ_B); u16* wk2p = (u16*)give(WSQ_B); u16* wv2p = (u16*)give(WSQ_B);
    u16* wo1p = (u16*)give(WSQ_B); u16* wo2p = (u16*)give(WSQ_B);
    u16* wf1p = (u16*)give(WFF_B); u16* wf2p = (u16*)give(WFF_B);
    u16* qb   = (u16*)give(ACT_B);
    u16* kb   = (u16*)give(ACT_B);
    u16* vb   = (u16*)give(ACT_B);
    u16* ab   = (u16*)give(ACT_B);
    float* t1 = (float*)give(ACT_F);
    float* n1f = (float*)give(ACT_F); u16* n1b = (u16*)give(ACT_B);
    float* n2f = (float*)give(ACT_F); u16* n2b = (u16*)give(ACT_B);
    u16* hb   = (u16*)give(HID_B);

    const float scale = 1.0f / sqrtf((float)DIM);
    const dim3 blk128(128);
    const dim3 gD (DIM / 64, ROWS / 128);  // N=512 GEMMs, 128-row tiles
    const dim3 gF1(FFD / 64, ROWS / 128);  // N=2048 GEMM
    const dim3 gAT(SEQ / 64, BSZ * NH);    // flash attention

    // --- weight repack / conversion (bf16) ---
    {
        const int nq = NH * DIM * HDM;
        repack_qkv_w<<<(nq + 255) / 256, 256, 0, stream>>>(Wq1, wq1p);
        repack_qkv_w<<<(nq + 255) / 256, 256, 0, stream>>>(Wk1, wk1p);
        repack_qkv_w<<<(nq + 255) / 256, 256, 0, stream>>>(Wv1, wv1p);
        repack_qkv_w<<<(nq + 255) / 256, 256, 0, stream>>>(Wq2, wq2p);
        repack_qkv_w<<<(nq + 255) / 256, 256, 0, stream>>>(Wk2, wk2p);
        repack_qkv_w<<<(nq + 255) / 256, 256, 0, stream>>>(Wv2, wv2p);
        cvt_f32_bf16<<<(DIM * DIM + 255) / 256, 256, 0, stream>>>(Wo1, wo1p, DIM * DIM);
        cvt_f32_bf16<<<(DIM * DIM + 255) / 256, 256, 0, stream>>>(Wo2, wo2p, DIM * DIM);
        cvt_f32_bf16<<<(DIM * FFD + 255) / 256, 256, 0, stream>>>(Wf1, wf1p, DIM * FFD);
        cvt_f32_bf16<<<(FFD * DIM + 255) / 256, 256, 0, stream>>>(Wf2, wf2p, FFD * DIM);
        cvt_f32_bf16<<<(ROWS * DIM + 255) / 256, 256, 0, stream>>>(x, xb, ROWS * DIM);
    }

    // --- MHA 1 (masked self-attention) ---
    gemm_bf16<false,false,false,true><<<gD, blk128, 0, stream>>>(xb, wq1p, bq1, nullptr, nullptr, qb, ROWS, DIM, DIM);
    gemm_bf16<false,false,false,true><<<gD, blk128, 0, stream>>>(xb, wk1p, bk1, nullptr, nullptr, kb, ROWS, DIM, DIM);
    gemm_bf16<false,false,false,true><<<gD, blk128, 0, stream>>>(xb, wv1p, bv1, nullptr, nullptr, vb, ROWS, DIM, DIM);
    flash_attn<<<gAT, blk128, 0, stream>>>(qb, kb, vb, tmask, ab, scale, 1);
    gemm_bf16<false,true,true,false><<<gD, blk128, 0, stream>>>(ab, wo1p, bo1, x, t1, nullptr, ROWS, DIM, DIM);
    layernorm_k<<<ROWS, blk128, 0, stream>>>(t1, g1, b1, n1f, n1b);

    // --- MHA 2 (unmasked, on n1) ---
    gemm_bf16<false,false,false,true><<<gD, blk128, 0, stream>>>(n1b, wq2p, bq2, nullptr, nullptr, qb, ROWS, DIM, DIM);
    gemm_bf16<false,false,false,true><<<gD, blk128, 0, stream>>>(n1b, wk2p, bk2, nullptr, nullptr, kb, ROWS, DIM, DIM);
    gemm_bf16<false,false,false,true><<<gD, blk128, 0, stream>>>(n1b, wv2p, bv2, nullptr, nullptr, vb, ROWS, DIM, DIM);
    flash_attn<<<gAT, blk128, 0, stream>>>(qb, kb, vb, nullptr, ab, scale, 0);
    gemm_bf16<false,true,true,false><<<gD, blk128, 0, stream>>>(ab, wo2p, bo2, n1f, t1, nullptr, ROWS, DIM, DIM);
    layernorm_k<<<ROWS, blk128, 0, stream>>>(t1, g2, b2, n2f, n2b);

    // --- FFN ---
    gemm_bf16<true,false,false,true><<<gF1, blk128, 0, stream>>>(n2b, wf1p, bf1, nullptr, nullptr, hb, ROWS, FFD, DIM);
    gemm_bf16<false,true,true,false><<<gD,  blk128, 0, stream>>>(hb, wf2p, bf2, n2f, t1, nullptr, ROWS, DIM, FFD);
    layernorm_k<<<ROWS, blk128, 0, stream>>>(t1, g3, b3, (float*)d_out, nullptr);
}
